// MultiHeadAttention_84610855731691
// MI455X (gfx1250) — matrix operand order repaired and checked by reading, not compiled, not run on hardware
//
#include <hip/hip_runtime.h>
#include <hip/hip_bf16.h>
#include <math.h>

// ---------------- problem constants ----------------
constexpr int B  = 2;
constexpr int S  = 2048;
constexpr int D  = 768;
constexpr int H  = 12;
constexpr int DK = 64;                     // D / H
constexpr int BS = B * S;                  // 4096 rows in the projection GEMMs
constexpr int NH_ELEMS = B * H * S * DK;   // elems of one head-split tensor (f16)
constexpr int LDSTRIDE = S + 16;           // 2064 ; 2064 % 64 == 16 -> rows 0..3
                                           // land on distinct bank groups
typedef _Float16 f16;
typedef __attribute__((ext_vector_type(16))) _Float16 v16h;
typedef __attribute__((ext_vector_type(8)))  float    v8f;

// ---------------- WMMA fragment helpers (contiguous k order, same for A and B) ------
// A-frag (16x32 f16): lane l holds row m = l&15; with hi = l>>4,
//   elem i in [0,15] <-> K = k0 + 16*hi + i
__device__ __forceinline__ v16h a_frag_f16(const f16* row, int k0, int hi) {
  const f16* p0 = row + k0 + 16 * hi;
  const f16* p1 = p0 + 8;
  v16h a;
#pragma unroll
  for (int i = 0; i < 8; ++i) { a[i] = p0[i]; a[8 + i] = p1[i]; }
  return a;
}
__device__ __forceinline__ v16h a_frag_cvt_f32(const float* row, int k0, int hi) {
  const float* p0 = row + k0 + 16 * hi;
  const float* p1 = p0 + 8;
  v16h a;
#pragma unroll
  for (int i = 0; i < 8; ++i) { a[i] = (f16)p0[i]; a[8 + i] = (f16)p1[i]; }
  return a;
}
// B-frag (32x16 f16): lane l holds column n = l&15; elem i <-> K = k0+16*hi+i.
// Caller passes a pointer where K is the unit stride => 16 contiguous halfs.
__device__ __forceinline__ v16h b_frag_contig_f16(const f16* p) {
  v16h b;
#pragma unroll
  for (int i = 0; i < 16; ++i) b[i] = p[i];
  return b;
}
__device__ __forceinline__ v16h b_frag_cvt_f32(const float* p) {
  v16h b;
#pragma unroll
  for (int i = 0; i < 16; ++i) b[i] = (f16)p[i];
  return b;
}
__device__ __forceinline__ v8f wmma32(v16h a, v16h b, v8f c) {
  return __builtin_amdgcn_wmma_f32_16x16x32_f16(false, a, false, b,
                                                (short)0, c, false, false);
}

// ---------------- kernel 1: fused QKV projections ---------------------------
// grid (BS/16, D/64, 3), block 32 (one wave). z: 0=Q, 1=K, 2=V.
// Each wave computes a 16x64 output tile (4 accumulators, shared A-frag):
//   out[row,n] = dot(X[row,:], W[n,:]) + b[n]     (X @ W^T + b)
// Q,K stored f16 [B,H,S,DK]; V stored f16 transposed [B,H,DK,S].
__global__ void proj_qkv(const float* __restrict__ q_in,
                         const float* __restrict__ k_in,
                         const float* __restrict__ v_in,
                         const float* __restrict__ Wq, const float* __restrict__ bq,
                         const float* __restrict__ Wk, const float* __restrict__ bk,
                         const float* __restrict__ Wv, const float* __restrict__ bv,
                         f16* __restrict__ qH, f16* __restrict__ kH,
                         f16* __restrict__ vT) {
  const int lane = threadIdx.x;
  const int lo = lane & 15, hi = lane >> 4;
  const int mt = blockIdx.x, nt = blockIdx.y, which = blockIdx.z;

  const float* X  = (which == 0) ? q_in : (which == 1) ? k_in : v_in;
  const float* W  = (which == 0) ? Wq   : (which == 1) ? Wk   : Wv;
  const float* bb = (which == 0) ? bq   : (which == 1) ? bk   : bv;

  const float* arow = X + (size_t)(mt * 16 + lo) * D;   // A row m = lo
  const float* brow[4];
#pragma unroll
  for (int j = 0; j < 4; ++j)
    brow[j] = W + (size_t)(nt * 64 + j * 16 + lo) * D;  // B col n = W row n

  v8f acc[4] = {};
#pragma unroll 2
  for (int k0 = 0; k0 < D; k0 += 32) {
    const v16h a = a_frag_cvt_f32(arow, k0, hi);        // shared across 4 tiles
#pragma unroll
    for (int j = 0; j < 4; ++j) {
      const v16h b = b_frag_cvt_f32(brow[j] + k0 + 16 * hi);
      acc[j] = wmma32(a, b, acc[j]);
    }
  }

#pragma unroll
  for (int j = 0; j < 4; ++j) {
    const int n_abs = nt * 64 + j * 16 + lo;  // C-tile column N = lo
    const float bn = bb[n_abs];
    const int h = n_abs >> 6, dk = n_abs & 63;
    if (which < 2) {                          // Q/K: [B,H,S,DK]
      f16* dst = (which == 0) ? qH : kH;
#pragma unroll
      for (int r = 0; r < 8; ++r) {
        const int M = 8 * hi + r;
        const int rg = mt * 16 + M;           // global row over B*S
        const int b_ = rg >> 11, s_ = rg & (S - 1);
        dst[(((size_t)(b_ * H + h) * S) + s_) * DK + dk] = (f16)(acc[j][r] + bn);
      }
    } else {                                  // V: transposed [B,H,DK,S]
#pragma unroll
      for (int r = 0; r < 8; ++r) {
        const int M = 8 * hi + r;
        const int rg = mt * 16 + M;
        const int b_ = rg >> 11, s_ = rg & (S - 1);
        vT[(((size_t)(b_ * H + h) * DK) + dk) * S + s_] = (f16)(acc[j][r] + bn);
      }
    }
  }
}

// ---------------- kernel 2: scores -> softmax -> attn out + ctx -------------
// grid (S/16, B*H), block 128 (4 waves). Dynamic LDS: 16 * LDSTRIDE floats.
// Wave w: scores for k-tile chunk w, then ctx output column block ns = w.
__global__ void attn_kernel(const f16* __restrict__ qH,
                            const f16* __restrict__ kH,
                            const f16* __restrict__ vT,
                            const int* __restrict__ mask,
                            float* __restrict__ attn_out,
                            f16* __restrict__ ctx) {
  extern __shared__ float sc[];              // [16][LDSTRIDE] score stripe
  __shared__ float pmx[4][16];               // per-wave row maxima
  __shared__ float psum[16][8];              // per-(row,segment) exp sums
  __shared__ float rmax[16];                 // row max
  __shared__ float rinv[16];                 // 1 / row sum

  const int tid  = threadIdx.x;
  const int w    = tid >> 5;                 // wave id 0..3
  const int lane = tid & 31;
  const int lo = lane & 15, hi = lane >> 4;
  const int qt = blockIdx.x;                 // 16-row query tile
  const int bh = blockIdx.y;
  const int b_ = bh / H, h_ = bh % H;

  const f16* Qrow = qH + ((size_t)bh * S + qt * 16 + lo) * DK;  // A row m = lo
  const f16* Kb   = kH + (size_t)bh * S * DK;
  const int* mrow = mask + (size_t)b_ * S * S + (size_t)(qt * 16) * S;

  // Q fragments are loop-invariant: hoist both K-steps (DK=64 => k0 = 0, 32).
  const v16h qa0 = a_frag_f16(Qrow, 0, hi);
  const v16h qa1 = a_frag_f16(Qrow, 32, hi);

  float pmax[8];
#pragma unroll
  for (int r = 0; r < 8; ++r) pmax[r] = -3.0e38f;

  // ---- scores: each wave owns 32 of the 128 k-tiles ----
  const int kt0 = w * (S / 16 / 4), kt1 = kt0 + (S / 16 / 4);
  for (int kt = kt0; kt < kt1; ++kt) {
    __builtin_prefetch(mrow + (8 * hi) * S + (kt + 1) * 16 + lo, 0, 1);
    v8f acc = {};
    acc = wmma32(qa0, b_frag_contig_f16(
                     Kb + (size_t)(kt * 16 + lo) * DK + 0 + 16 * hi), acc);
    acc = wmma32(qa1, b_frag_contig_f16(
                     Kb + (size_t)(kt * 16 + lo) * DK + 32 + 16 * hi), acc);
    const int kcol = kt * 16 + lo;           // C-tile: N = lo, M = 8*hi + r
#pragma unroll
    for (int r = 0; r < 8; ++r) {
      const int M = 8 * hi + r;
      const int mv = mrow[(size_t)M * S + kcol];
      const float v = mv ? acc[r] * 0.125f : -1.0e9f;   // 1/sqrt(64)
      sc[M * LDSTRIDE + kcol] = v;
      pmax[r] = fmaxf(pmax[r], v);
    }
  }

  // ---- row max: shfl-reduce across the 16 lanes sharing hi, then LDS ----
#pragma unroll
  for (int off = 1; off < 16; off <<= 1)
#pragma unroll
    for (int r = 0; r < 8; ++r)
      pmax[r] = fmaxf(pmax[r], __shfl_xor(pmax[r], off, 32));
  if (lo == 0)
#pragma unroll
    for (int r = 0; r < 8; ++r) pmx[w][8 * hi + r] = pmax[r];
  __syncthreads();
  if (tid < 16)
    rmax[tid] = fmaxf(fmaxf(pmx[0][tid], pmx[1][tid]),
                      fmaxf(pmx[2][tid], pmx[3][tid]));
  __syncthreads();

  // ---- exp + partial sums: thread = (row = tid>>3, segment = tid&7),
  //      stride-8 columns so the 8 threads of a row hit distinct banks ----
  {
    const int row = tid >> 3, seg = tid & 7;
    const float rm = rmax[row];
    float* rowp = sc + row * LDSTRIDE;
    float sum = 0.f;
    for (int c = seg; c < S; c += 8) {
      const float e = __expf(rowp[c] - rm);
      rowp[c] = e;                           // keep unnormalized exp in LDS
      sum += e;
    }
    psum[row][seg] = sum;
  }
  __syncthreads();
  if (tid < 16) {
    float s = 0.f;
#pragma unroll
    for (int j = 0; j < 8; ++j) s += psum[tid][j];
    rinv[tid] = 1.0f / s;
  }
  __syncthreads();

  // ---- write normalized attn: 32768 elements, 128 threads, coalesced ----
  float* aout = attn_out + ((size_t)bh * S + qt * 16) * S;
  for (int idx = tid; idx < 16 * S; idx += 128) {
    const int r = idx >> 11, c = idx & (S - 1);
    aout[(size_t)r * S + c] = sc[r * LDSTRIDE + c] * rinv[r];
  }

  // ---- ctx = attn @ V : wave w computes column block ns = w (16 of DK=64),
  //      folding 1/rowsum into the epilogue ----
  const f16* Vb = vT + (size_t)bh * DK * S;
  {
    const int ns = w;
    v8f acc = {};
    const float* arow = sc + lo * LDSTRIDE;  // A row m = lo (f32 exp in LDS)
    for (int k0 = 0; k0 < S; k0 += 32) {
      v16h a;
      {
        const float* p0 = arow + k0 + 16 * hi;
        const float* p1 = p0 + 8;
#pragma unroll
        for (int i = 0; i < 8; ++i) { a[i] = (f16)p0[i]; a[8 + i] = (f16)p1[i]; }
      }
      const v16h b = b_frag_contig_f16(
          Vb + (size_t)(ns * 16 + lo) * S + k0 + 16 * hi);
      acc = wmma32(a, b, acc);
    }
#pragma unroll
    for (int r = 0; r < 8; ++r) {
      const int M = 8 * hi + r;
      const float val = acc[r] * rinv[M];
      const int s_ = qt * 16 + M;
      const int d_ = h_ * DK + ns * 16 + lo;
      ctx[((size_t)b_ * S + s_) * D + d_] = (f16)val;
    }
  }
}

// ---------------- kernel 3: output projection --------------------------------
// grid (BS/16, D/64), block 32. out = ctx @ Wo^T + bo (f32), 16x64 per wave.
__global__ void out_proj(const f16* __restrict__ ctx,
                         const float* __restrict__ Wo,
                         const float* __restrict__ bo,
                         float* __restrict__ out) {
  const int lane = threadIdx.x;
  const int lo = lane & 15, hi = lane >> 4;
  const int mt = blockIdx.x, nt = blockIdx.y;

  const f16* arow = ctx + (size_t)(mt * 16 + lo) * D;
  const float* brow[4];
#pragma unroll
  for (int j = 0; j < 4; ++j)
    brow[j] = Wo + (size_t)(nt * 64 + j * 16 + lo) * D;

  v8f acc[4] = {};
#pragma unroll 2
  for (int k0 = 0; k0 < D; k0 += 32) {
    const v16h a = a_frag_f16(arow, k0, hi);
#pragma unroll
    for (int j = 0; j < 4; ++j) {
      const v16h b = b_frag_cvt_f32(brow[j] + k0 + 16 * hi);
      acc[j] = wmma32(a, b, acc[j]);
    }
  }
#pragma unroll
  for (int j = 0; j < 4; ++j) {
    const float bn = bo[nt * 64 + j * 16 + lo];
#pragma unroll
    for (int r = 0; r < 8; ++r) {
      const int M = 8 * hi + r;
      out[(size_t)(mt * 16 + M) * D + nt * 64 + j * 16 + lo] = acc[j][r] + bn;
    }
  }
}

// ---------------- launch -----------------------------------------------------
extern "C" void kernel_launch(void* const* d_in, const int* in_sizes, int n_in,
                              void* d_out, int out_size, void* d_ws, size_t ws_size,
                              hipStream_t stream) {
  const float* query = (const float*)d_in[0];
  const float* key_  = (const float*)d_in[1];
  const float* value = (const float*)d_in[2];
  const int*   mask  = (const int*)d_in[3];
  const float* Wq = (const float*)d_in[4];  const float* bq = (const float*)d_in[5];
  const float* Wk = (const float*)d_in[6];  const float* bk = (const float*)d_in[7];
  const float* Wv = (const float*)d_in[8];  const float* bv = (const float*)d_in[9];
  const float* Wo = (const float*)d_in[10]; const float* bo = (const float*)d_in[11];

  float* out  = (float*)d_out;                       // [B,S,D]
  float* attn = out + (size_t)B * S * D;             // [B,H,S,S]

  f16* qH  = (f16*)d_ws;                             // [B,H,S,DK] f16
  f16* kH  = qH + (size_t)NH_ELEMS;                  // [B,H,S,DK] f16
  f16* vT  = kH + (size_t)NH_ELEMS;                  // [B,H,DK,S] f16
  f16* ctx = vT + (size_t)NH_ELEMS;                  // [B,S,D]    f16

  proj_qkv<<<dim3(BS / 16, D / 64, 3), dim3(32), 0, stream>>>(
      query, key_, value, Wq, bq, Wk, bk, Wv, bv, qH, kH, vT);

  const size_t lds_bytes = (size_t)16 * LDSTRIDE * sizeof(float);  // ~129 KB
  attn_kernel<<<dim3(S / 16, B * H), dim3(128), lds_bytes, stream>>>(
      qH, kH, vT, mask, attn, ctx);

  out_proj<<<dim3(BS / 16, D / 64), dim3(32), 0, stream>>>(ctx, Wo, bo, out);
}
